// DockPointNet_55688545960611
// MI455X (gfx1250) — matrix operand order repaired and checked by reading, not compiled, not run on hardware
//
#include <hip/hip_runtime.h>

// ---------------------------------------------------------------------------
// DockPointNet forward for MI455X (gfx1250, wave32, WMMA 16x16x32 f16).
//  - 16 rows per wave per WMMA tile, fp16 A/B, fp32 accumulate
//  - native v_tanh_f32 (CDNA5 TRANS op)
//  - segment-max via order-preserving int keys -> single global_atomic_max_i32
//  - ALL B operands pre-packed into WMMA fragment order (32B/lane, L2-resident)
//  - x-feature gather via GLOBAL_LOAD_ASYNC_TO_LDS (ASYNCcnt) when available
// ---------------------------------------------------------------------------

typedef __attribute__((ext_vector_type(16))) _Float16 v16h;
typedef __attribute__((ext_vector_type(8)))  _Float16 v8h;
typedef __attribute__((ext_vector_type(8)))  float    v8f;

#define NATOM 100000
#define NEDGE 1000000
#define NRES  10000
#define NPAIR 50000
#define EPSF  1e-12f
#define ENC_NEG1 0xC07FFFFF   // ordered-key encoding of -1.0f (segmax identity)

#define AS1 __attribute__((address_space(1)))
#define AS3 __attribute__((address_space(3)))

#define WMMA_F16(A, B, C) \
  __builtin_amdgcn_wmma_f32_16x16x32_f16(false, (A), false, (B), (short)0, (C), false, false)

#if defined(__has_builtin)
#  if __has_builtin(__builtin_amdgcn_global_load_async_to_lds_b32)
#    define HAVE_ASYNC_LDS 1
#  endif
#  if __has_builtin(__builtin_amdgcn_s_wait_asynccnt)
#    define ASYNC_WAIT0() __builtin_amdgcn_s_wait_asynccnt(0)
#  endif
#endif
#ifndef HAVE_ASYNC_LDS
#  define HAVE_ASYNC_LDS 0
#endif
#ifndef ASYNC_WAIT0
#  define ASYNC_WAIT0() asm volatile("s_wait_asynccnt 0x0" ::: "memory")
#endif

// Native CDNA5 transcendental tanh; v_nop covers the TRANS32 result hazard.
__device__ __forceinline__ float tanh_fast(float x) {
  float r;
  asm volatile("v_tanh_f32 %0, %1\n\tv_nop" : "=v"(r) : "v"(x));
  return r;
}

// Order-preserving float<->int bijection (involution): int compare == float
// compare, so segment-max is one uniform global_atomic_max_i32.
__device__ __forceinline__ int ordkey(int s) { return s ^ ((s >> 31) & 0x7fffffff); }
__device__ __forceinline__ int enc_f(float f) { return ordkey(__float_as_int(f)); }
__device__ __forceinline__ float dec_k(int k) { return __int_as_float(ordkey(k)); }

// sin/cos of atan2(||a x b||, a.b) without trig.
__device__ __forceinline__ void angle_sincos(float ax, float ay, float az,
                                             float bx, float by, float bz,
                                             float* s, float* c) {
  float cx = ay * bz - az * by;
  float cy = az * bx - ax * bz;
  float cz = ax * by - ay * bx;
  float sv = sqrtf(cx * cx + cy * cy + cz * cz + EPSF);
  float cv = ax * bx + ay * by + az * bz;
  float inv = rsqrtf(sv * sv + cv * cv);
  *s = sv * inv;
  *c = cv * inv;
}

__global__ void init_kernel(int* buf, long n, int val) {
  long i = (long)blockIdx.x * blockDim.x + threadIdx.x;
  long stride = (long)gridDim.x * blockDim.x;
  for (; i < n; i += stride) buf[i] = val;
}

// Pack W [K=ktiles*32][ncols] (f32 row-major) into WMMA-B fragment order:
// out[((n*ktiles + kt)*32 + lane)*16 + e] so each lane's fragment is 32B.
__global__ void pack_b_kernel(const float* __restrict__ W, _Float16* __restrict__ out,
                              int ktiles, int ncols, int total) {
  const int idx = blockIdx.x * blockDim.x + threadIdx.x;
  if (idx >= total) return;
  const int lane = idx & 31;
  const int kt = (idx >> 5) % ktiles;
  const int n  = (idx >> 5) / ktiles;
  const int koff = (lane >> 4) << 3;
  const int col  = lane & 15;
  #pragma unroll
  for (int e = 0; e < 16; ++e)
    out[(long)idx * 16 + e] = (_Float16)W[(long)(kt * 32 + koff + e + (e >= 8 ? 8 : 0)) * ncols + n * 16 + col];
}

// Build an A fragment (one 32-wide k-tile) from a f32 LDS row.
__device__ __forceinline__ v16h frag_from_f32(const float* rowbase, int khalf) {
  const float4* p = (const float4*)(rowbase + khalf);
  float tmp[16];
  *(float4*)&tmp[0]  = p[0];
  *(float4*)&tmp[4]  = p[1];
  *(float4*)&tmp[8]  = p[4];   // +16 floats (k offset 16)
  *(float4*)&tmp[12] = p[5];
  v16h a;
  #pragma unroll
  for (int e = 0; e < 16; ++e) a[e] = (_Float16)tmp[e];
  return a;
}

// ---------------------------------------------------------------------------
// Edge conv: msg[16x32] -> tanh(@W1+b1) -> tanh(@W2+b2) -> atomic segmax keys.
// 8 waves/block, 16 edges/wave. Weights pre-packed in global (L2-resident).
// ---------------------------------------------------------------------------
__global__ __launch_bounds__(256)
void ppf_conv_kernel(const float* __restrict__ x, const float* __restrict__ pos,
                     const float* __restrict__ nrm, const int* __restrict__ ei,
                     const _Float16* __restrict__ pW1, const float* __restrict__ b1,
                     const _Float16* __restrict__ pW2, const float* __restrict__ b2,
                     int* __restrict__ outmax, float inv_radius) {
  __shared__ __align__(32) float    sMsg[8][16][32];   // per-wave [row][k], f32
  __shared__ __align__(32) _Float16 sH1[8][16][32];
  __shared__ int sTgt[8][16];

  const int t = threadIdx.x;
  const int wave = t >> 5;
  const int lane = t & 31;
  const int row  = lane & 15;
  const int hi   = lane >> 4;
  const int khalf = hi * 8;
  const int col   = row;

  const long tileBase = ((long)blockIdx.x * 8 + wave) * 16;
  if (lane < 16) {
    long e = tileBase + lane;
    if (e >= NEDGE) e = NEDGE - 1;             // duplicate edge: idempotent under max
    const int j = ei[e];                       // source
    const int i = ei[NEDGE + e];               // target
    sTgt[wave][lane] = i;

    float* dstRow = &sMsg[wave][lane][0];
    float* srcRow = const_cast<float*>(x) + (long)j * 25;
#if HAVE_ASYNC_LDS
    // DMA the 25 feature dwords straight into LDS (per-lane addr pairs).
    #pragma unroll
    for (int c = 0; c < 25; ++c)
      __builtin_amdgcn_global_load_async_to_lds_b32(
          (AS1 int*)(srcRow + c), (AS3 int*)(dstRow + c), 0, 0);
#else
    #pragma unroll
    for (int c = 0; c < 25; ++c) dstRow[c] = srcRow[c];
#endif

    const float dx = pos[3 * j + 0] - pos[3 * i + 0];
    const float dy = pos[3 * j + 1] - pos[3 * i + 1];
    const float dz = pos[3 * j + 2] - pos[3 * i + 2];
    const float dist = sqrtf(dx * dx + dy * dy + dz * dz + EPSF);
    const float nix = nrm[3 * i + 0], niy = nrm[3 * i + 1], niz = nrm[3 * i + 2];
    const float njx = nrm[3 * j + 0], njy = nrm[3 * j + 1], njz = nrm[3 * j + 2];
    float s1, c1, s2, c2, s3, c3;
    angle_sincos(nix, niy, niz, dx, dy, dz, &s1, &c1);
    angle_sincos(njx, njy, njz, dx, dy, dz, &s2, &c2);
    angle_sincos(nix, niy, niz, njx, njy, njz, &s3, &c3);

    dstRow[25] = dist * inv_radius;
    dstRow[26] = s1;  dstRow[27] = c1;
    dstRow[28] = s2;  dstRow[29] = c2;
    dstRow[30] = s3;  dstRow[31] = c3;
  }
  ASYNC_WAIT0();        // our async LDS writes are complete before we signal
  __syncthreads();

  const v16h a = frag_from_f32(&sMsg[wave][row][0], khalf);

  const v16h* W1v = (const v16h*)pW1;
  const v16h* W2v = (const v16h*)pW2;

  // Layer 1: 32x32 weights = two 16-col tiles.
  #pragma unroll
  for (int n = 0; n < 2; ++n) {
    v8f c = {};
    c = WMMA_F16(a, W1v[n * 32 + lane], c);
    const float bias = b1[n * 16 + col];
    #pragma unroll
    for (int r = 0; r < 8; ++r)
      sH1[wave][r + hi * 8][n * 16 + col] = (_Float16)tanh_fast(c[r] + bias);
  }
  __syncthreads();

  union { v16h v; v8h h[2]; } ua2;
  ua2.h[0] = *(const v8h*)&sH1[wave][row][khalf];
  ua2.h[1] = *(const v8h*)&sH1[wave][row][16 + khalf];
  const v16h a2 = ua2.v;

  // Layer 2: 32x128 = eight 16-col tiles; fused tanh + segment-max.
  #pragma unroll
  for (int n = 0; n < 8; ++n) {
    v8f c = {};
    c = WMMA_F16(a2, W2v[n * 32 + lane], c);
    const float bias = b2[n * 16 + col];
    #pragma unroll
    for (int r = 0; r < 8; ++r) {
      const int tgt = sTgt[wave][r + hi * 8];
      atomicMax(&outmax[(long)tgt * 128 + n * 16 + col],
                enc_f(tanh_fast(c[r] + bias)));
    }
  }
}

// Load one WMMA A-fragment k-tile (32 wide) from an ordered-key int buffer.
__device__ __forceinline__ v16h load_a_keys(const int* base, int khalf) {
  const int4* pr = (const int4*)(base + khalf);
  int tmp[16];
  *(int4*)&tmp[0]  = pr[0];
  *(int4*)&tmp[4]  = pr[1];
  *(int4*)&tmp[8]  = pr[4];   // +16 ints (k offset 16)
  *(int4*)&tmp[12] = pr[5];
  v16h a;
  #pragma unroll
  for (int e = 0; e < 16; ++e) a[e] = (_Float16)dec_k(tmp[e]);
  return a;
}

// ---------------------------------------------------------------------------
// Atom MLP [N,384]@[384,256] + tanh, fused residue segment-max (int keys).
// ---------------------------------------------------------------------------
__global__ __launch_bounds__(256)
void atom_kernel(const int* __restrict__ o2, const int* __restrict__ o3,
                 const int* __restrict__ o4, const _Float16* __restrict__ pWa,
                 const float* __restrict__ ba, const int* __restrict__ atom2res,
                 int* __restrict__ resmax) {
  const int t = threadIdx.x;
  const int wave = t >> 5, lane = t & 31;
  const int row = lane & 15, hi = lane >> 4;
  const int khalf = hi * 8, col = row;
  const long rowBase = ((long)blockIdx.x * 8 + wave) * 16;
  long myrow = rowBase + row;
  if (myrow >= NATOM) myrow = NATOM - 1;

  v16h aF[12];
  #pragma unroll
  for (int kt = 0; kt < 12; ++kt) {
    const int* bp = (kt < 4) ? o2 : (kt < 8 ? o3 : o4);
    aF[kt] = load_a_keys(bp + myrow * 128 + (kt & 3) * 32, khalf);
  }

  const v16h* Wv = (const v16h*)pWa;
  for (int n = 0; n < 16; ++n) {   // 256 cols = 16 tiles
    v8f c = {};
    #pragma unroll
    for (int kt = 0; kt < 12; ++kt)
      c = WMMA_F16(aF[kt], Wv[(n * 12 + kt) * 32 + lane], c);
    const float bias = ba[n * 16 + col];
    #pragma unroll
    for (int r = 0; r < 8; ++r) {
      long rr = rowBase + r + hi * 8;
      if (rr >= NATOM) rr = NATOM - 1;        // duplicate atom: idempotent
      const int res = atom2res[rr];
      atomicMax(&resmax[(long)res * 256 + n * 16 + col],
                enc_f(tanh_fast(c[r] + bias)));
    }
  }
}

// ---------------------------------------------------------------------------
// Residue MLP [R,256]@[256,512] + tanh -> res512 (plain f32).
// ---------------------------------------------------------------------------
__global__ __launch_bounds__(256)
void res_kernel(const int* __restrict__ resmax, const _Float16* __restrict__ pWr,
                const float* __restrict__ br, float* __restrict__ res512) {
  const int t = threadIdx.x;
  const int wave = t >> 5, lane = t & 31;
  const int row = lane & 15, hi = lane >> 4;
  const int khalf = hi * 8, col = row;
  const long rowBase = ((long)blockIdx.x * 8 + wave) * 16;
  long myrow = rowBase + row;
  if (myrow >= NRES) myrow = NRES - 1;

  v16h aF[8];
  #pragma unroll
  for (int kt = 0; kt < 8; ++kt)
    aF[kt] = load_a_keys(resmax + myrow * 256 + kt * 32, khalf);

  const v16h* Wv = (const v16h*)pWr;
  for (int n = 0; n < 32; ++n) {   // 512 cols = 32 tiles
    v8f c = {};
    #pragma unroll
    for (int kt = 0; kt < 8; ++kt)
      c = WMMA_F16(aF[kt], Wv[(n * 8 + kt) * 32 + lane], c);
    const float bias = br[n * 16 + col];
    #pragma unroll
    for (int r = 0; r < 8; ++r) {
      long rr = rowBase + r + hi * 8;
      if (rr >= NRES) rr = NRES - 1;          // duplicate row: same value
      res512[rr * 512 + n * 16 + col] = tanh_fast(c[r] + bias);
    }
  }
}

// ---------------------------------------------------------------------------
// Final: out[p] = (res512[src]-res512[tgt]) . Wl + bl
// ---------------------------------------------------------------------------
__global__ __launch_bounds__(256)
void pair_kernel(const float* __restrict__ res512, const int* __restrict__ src,
                 const int* __restrict__ tgt, const float* __restrict__ Wl,
                 const float* __restrict__ bl, float* __restrict__ out) {
  const int p = blockIdx.x * blockDim.x + threadIdx.x;
  if (p >= NPAIR) return;
  const float4* a = (const float4*)(res512 + (long)src[p] * 512);
  const float4* b = (const float4*)(res512 + (long)tgt[p] * 512);
  const float4* w = (const float4*)Wl;
  float acc = 0.0f;
  #pragma unroll 4
  for (int k = 0; k < 128; ++k) {
    const float4 av = a[k], bv = b[k], wv = w[k];
    acc += (av.x - bv.x) * wv.x + (av.y - bv.y) * wv.y +
           (av.z - bv.z) * wv.z + (av.w - bv.w) * wv.w;
  }
  out[p] = acc + bl[0];
}

extern "C" void kernel_launch(void* const* d_in, const int* in_sizes, int n_in,
                              void* d_out, int out_size, void* d_ws, size_t ws_size,
                              hipStream_t stream) {
  const float* x    = (const float*)d_in[0];
  const float* pos  = (const float*)d_in[1];
  const float* nrm  = (const float*)d_in[2];
  const int*   ei2  = (const int*)d_in[3];
  const int*   ei3  = (const int*)d_in[4];
  const int*   ei4  = (const int*)d_in[5];
  const int*   a2r  = (const int*)d_in[6];
  const int*   srci = (const int*)d_in[7];
  const int*   tgti = (const int*)d_in[8];
  const float* cW1[3] = {(const float*)d_in[9],  (const float*)d_in[13], (const float*)d_in[17]};
  const float* cb1[3] = {(const float*)d_in[10], (const float*)d_in[14], (const float*)d_in[18]};
  const float* cW2[3] = {(const float*)d_in[11], (const float*)d_in[15], (const float*)d_in[19]};
  const float* cb2[3] = {(const float*)d_in[12], (const float*)d_in[16], (const float*)d_in[20]};
  const float* Wa   = (const float*)d_in[21];
  const float* ba   = (const float*)d_in[22];
  const float* Wr   = (const float*)d_in[23];
  const float* br   = (const float*)d_in[24];
  const float* Wl   = (const float*)d_in[25];
  const float* bl   = (const float*)d_in[26];
  const int* cei[3] = {ei2, ei3, ei4};
  const float inv_r[3] = {1.0f / 5.0f, 1.0f / 8.5f, 1.0f / 10.0f};

  // Workspace layout
  int*   o2     = (int*)d_ws;
  int*   o3     = o2 + 1L * NATOM * 128;
  int*   o4     = o2 + 2L * NATOM * 128;
  int*   obuf[3] = {o2, o3, o4};
  int*   resmax = o2 + 3L * NATOM * 128;
  float* res512 = (float*)(resmax + (long)NRES * 256);
  _Float16* pWa = (_Float16*)(res512 + (long)NRES * 512);   // 32B-aligned offsets
  _Float16* pWr = pWa + 98304;                              // 12*16*32*16
  _Float16* pCW[6];                                         // pW1 x3, pW2 x3
  {
    _Float16* p = pWr + 131072;                             // 32*8*32*16
    for (int k = 0; k < 3; ++k) { pCW[k] = p;     p += 2 * 32 * 16; }   // W1: 1024
    for (int k = 0; k < 3; ++k) { pCW[3 + k] = p; p += 8 * 32 * 16; }   // W2: 4096
  }

  // segment-max identities (ordered-key encoding of -1.0f)
  const long initN = 3L * NATOM * 128 + (long)NRES * 256;
  init_kernel<<<4096, 256, 0, stream>>>(o2, initN, (int)ENC_NEG1);

  // Pre-pack all GEMM B operands into fragment order.
  pack_b_kernel<<<24, 256, 0, stream>>>(Wa, pWa, 12, 256, 16 * 12 * 32);
  pack_b_kernel<<<32, 256, 0, stream>>>(Wr, pWr, 8, 512, 32 * 8 * 32);
  for (int k = 0; k < 3; ++k) {
    pack_b_kernel<<<1, 256, 0, stream>>>(cW1[k], pCW[k],     1, 32,  2 * 32);
    pack_b_kernel<<<1, 256, 0, stream>>>(cW2[k], pCW[3 + k], 1, 128, 8 * 32);
  }

  const int gridE = (NEDGE + 127) / 128;   // 8 waves x 16 edges per block
  for (int k = 0; k < 3; ++k)
    ppf_conv_kernel<<<gridE, 256, 0, stream>>>(x, pos, nrm, cei[k],
                                               pCW[k], cb1[k], pCW[3 + k], cb2[k],
                                               obuf[k], inv_r[k]);

  atom_kernel<<<(NATOM + 127) / 128, 256, 0, stream>>>(o2, o3, o4, pWa, ba, a2r, resmax);
  res_kernel<<<(NRES + 127) / 128, 256, 0, stream>>>(resmax, pWr, br, res512);
  pair_kernel<<<(NPAIR + 255) / 256, 256, 0, stream>>>(res512, srci, tgti, Wl, bl, (float*)d_out);
}
